// LibxLSTM_NNS_53412213293472
// MI455X (gfx1250) — matrix operand order repaired and checked by reading, not compiled, not run on hardware
//
#include <hip/hip_runtime.h>
#include <hip/hip_bf16.h>

typedef __attribute__((ext_vector_type(16))) _Float16 v16h;
typedef __attribute__((ext_vector_type(8)))  float    v8f;

constexpr int kNH  = 4;
constexpr int kHD  = 32;
constexpr int kE   = 128;
constexpr int kSEQ = 64;
constexpr int kB   = 4096;
constexpr float kEPS = 1e-5f;

// ---- branch-free transcendentals (keep EXEC untouched inside the WMMA scan loop) ----
__device__ __forceinline__ float fast_rcp(float x) {
#if __has_builtin(__builtin_amdgcn_rcpf)
  return __builtin_amdgcn_rcpf(x);
#else
  return 1.f / x;
#endif
}
__device__ __forceinline__ float fast_tanh(float x) {
#if __has_builtin(__builtin_amdgcn_tanhf)
  return __builtin_amdgcn_tanhf(x);          // gfx1250 v_tanh_f32 (TRANS unit)
#else
  float e = __expf(2.f * x);
  return 1.f - 2.f * fast_rcp(e + 1.f);
#endif
}
__device__ __forceinline__ float fast_sigmoid(float x) {
  return fast_rcp(1.f + __expf(-x));
}
__device__ __forceinline__ float log_sigmoid(float x) {
  return fminf(x, 0.f) - __logf(1.f + __expf(-fabsf(x)));
}
__device__ __forceinline__ float erf_approx(float x) {   // A&S 7.1.26, branchless, |err|<1.5e-7
  float ax = fabsf(x);
  float t  = fast_rcp(1.f + 0.3275911f * ax);
  float p  = t * (0.254829592f + t * (-0.284496736f +
             t * (1.421413741f + t * (-1.453152027f + t * 1.061405429f))));
  float r  = 1.f - p * __expf(-ax * ax);
  return copysignf(r, x);
}

// ---- WMMA fragment index helpers (CDNA5 16-bit layouts, CDNA5 ISA §7.12.2) ----
// A 16x32 f16: lanes0-15: vgpr0-3 -> K=0..7, vgpr4-7 -> K=16..23; lanes16-31: K=8..15 / 24..31
__device__ __forceinline__ int a_k(int lane, int e) {
  int j = e >> 1, p = e & 1;
  int base;
  if (lane < 16) base = (j < 4) ? (2 * j) : (16 + 2 * (j - 4));
  else           base = (j < 4) ? (8 + 2 * j) : (24 + 2 * (j - 4));
  return base + p;
}
// B 32x16 f16: same K order as A: lanes0-15 K=0..7,16..23; lanes16-31 K=8..15,24..31; N = lane%16
__device__ __forceinline__ int b_k(int lane, int e) {
  int h8 = (lane < 16) ? 0 : 8;
  return (e < 8) ? (h8 + e) : (16 + h8 + (e - 8));
}
// C/D f32 16x16: vgpr r -> M = r (lanes0-15) / 8+r (lanes16-31), N = lane%16

// =================== input projection: h = x @ w_in + b_in (D=3) ===================
__global__ void proj_kernel(const float* __restrict__ x, const float* __restrict__ w_in,
                            const float* __restrict__ b_in, float* __restrict__ h, int total) {
  int i = blockIdx.x * blockDim.x + threadIdx.x;
  if (i >= total) return;
  int e  = i & (kE - 1);
  int bs = i >> 7;
  float acc = b_in[e];
#pragma unroll
  for (int d = 0; d < 3; ++d) acc += x[bs * 3 + d] * w_in[d * kE + e];
  h[i] = acc;
}

// =================== sLSTM scan: LN1 + gates + recurrence + cell + groupnorm + residual ===================
__launch_bounds__(256)
__global__ void scan_kernel(float* __restrict__ h,
                            const float* __restrict__ wg,    // (4,NH,HD,HD)  [g][h][o][d]
                            const float* __restrict__ rk,    // (NH,HD,4,HD)  [h][d][g][k]
                            const float* __restrict__ bcell, // (4,NH,HD)
                            const float* __restrict__ ln1w,
                            const float* __restrict__ gnw) {
  __shared__ _Float16 sHN[16][kE + 8];
  __shared__ _Float16 sY [16][kE + 8];
  const int tid  = threadIdx.x;
  const int lane = tid & 31;
  const int w    = tid >> 5;
  const int head = w >> 1, nt = w & 1;
  const int r  = tid >> 4;        // row-parallel mapping: 16 lanes per row
  const int q  = tid & 15;
  const int c0 = q * 8;
  const int b0 = blockIdx.x * 16;

  // Preload B-fragments of combined weights [Wg ; R] into registers (resident for all 64 steps)
  v16h  wB[8];
  float bias[4];
  {
    const int N = nt * 16 + (lane & 15);
#pragma unroll
    for (int g = 0; g < 4; ++g) {
      v16h bi, br;
#pragma unroll
      for (int e = 0; e < 16; ++e) {
        int K = b_k(lane, e);
        bi[e] = (_Float16)wg[((g * kNH + head) * kHD + N) * kHD + K];   // gates[..o]=sum_d hn*wg[g,h,o,d]
        br[e] = (_Float16)rk[((head * kHD + K) * 4 + g) * kHD + N];     // Ry[..k]=sum_d y*rk[h,d,g,k]
      }
      wB[g * 2]     = bi;
      wB[g * 2 + 1] = br;
      bias[g] = bcell[(g * kNH + head) * kHD + N];
    }
  }
  float w1[8], wgn[8];
#pragma unroll
  for (int k = 0; k < 8; ++k) { w1[k] = ln1w[c0 + k]; wgn[k] = gnw[c0 + k]; }

  for (int i = tid; i < 16 * (kE + 8); i += 256) ((_Float16*)sY)[i] = (_Float16)0.f;
  float c_[8], n_[8], m_[8];
#pragma unroll
  for (int k = 0; k < 8; ++k) { c_[k] = 0.f; n_[k] = 0.f; m_[k] = 0.f; }
  __syncthreads();

  for (int t = 0; t < kSEQ; ++t) {
    float* hrow = h + ((size_t)(b0 + r) * kSEQ + t) * kE + c0;
    float hreg[8] __attribute__((aligned(16)));
    *(float4*)(hreg)     = *(const float4*)(hrow);
    *(float4*)(hreg + 4) = *(const float4*)(hrow + 4);
    if (t + 1 < kSEQ) __builtin_prefetch(hrow + kE, 0, 0);

    // LN1 over E=128 (16-lane row groups)
    float s1 = 0.f, s2 = 0.f;
#pragma unroll
    for (int k = 0; k < 8; ++k) { s1 += hreg[k]; s2 += hreg[k] * hreg[k]; }
#pragma unroll
    for (int off = 8; off >= 1; off >>= 1) { s1 += __shfl_xor(s1, off, 32); s2 += __shfl_xor(s2, off, 32); }
    float mu = s1 * (1.f / 128.f);
    float rs = rsqrtf(s2 * (1.f / 128.f) - mu * mu + kEPS);
#pragma unroll
    for (int k = 0; k < 8; ++k) sHN[r][c0 + k] = (_Float16)((hreg[k] - mu) * rs * w1[k]);
    __syncthreads();

    // Gather A fragments (hn_t and y_{t-1}) from LDS
    v16h aIn, aY;
#pragma unroll
    for (int e = 0; e < 16; ++e) {
      int K  = a_k(lane, e);
      aIn[e] = sHN[lane & 15][head * kHD + K];
      aY [e] = sY [lane & 15][head * kHD + K];
    }
    __syncthreads();

    // raw[g] = hn@Wg + y@R + b  (two K=32 WMMAs per gate)
    v8f raw[4];
#pragma unroll
    for (int g = 0; g < 4; ++g) {
      v8f acc;
#pragma unroll
      for (int k = 0; k < 8; ++k) acc[k] = bias[g];
      acc = __builtin_amdgcn_wmma_f32_16x16x32_f16(false, aIn, false, wB[g * 2],     (short)0, acc, false, false);
      acc = __builtin_amdgcn_wmma_f32_16x16x32_f16(false, aY,  false, wB[g * 2 + 1], (short)0, acc, false, false);
      raw[g] = acc;
    }

    // stabilized sLSTM cell, branch-free, all in f32 registers
    float ynew[8];
#pragma unroll
    for (int k = 0; k < 8; ++k) {
      float iraw = raw[0][k], fraw = raw[1][k], zraw = raw[2][k], oraw = raw[3][k];
      float lsf  = log_sigmoid(fraw);
      float lfm  = m_[k] + lsf;
      float mnew = (n_[k] == 0.f) ? iraw : fmaxf(iraw, lfm);
      float og   = fast_sigmoid(oraw);
      float ig   = __expf(iraw - mnew);
      float fg   = __expf(lfm - mnew);
      float cnew = fg * c_[k] + ig * fast_tanh(zraw);
      float nnew = fg * n_[k] + ig;
      c_[k] = cnew; n_[k] = nnew; m_[k] = mnew;
      ynew[k] = og * cnew * fast_rcp(nnew);
    }
    {
      int Mb  = (lane < 16) ? 0 : 8;
      int col = head * kHD + nt * 16 + (lane & 15);
#pragma unroll
      for (int k = 0; k < 8; ++k) sY[Mb + k][col] = (_Float16)ynew[k];
    }
    __syncthreads();

    // per-head group norm of y + residual into h
    float yv[8], g1 = 0.f, g2 = 0.f;
#pragma unroll
    for (int k = 0; k < 8; ++k) { yv[k] = (float)sY[r][c0 + k]; g1 += yv[k]; g2 += yv[k] * yv[k]; }
    g1 += __shfl_xor(g1, 1, 32); g2 += __shfl_xor(g2, 1, 32);
    g1 += __shfl_xor(g1, 2, 32); g2 += __shfl_xor(g2, 2, 32);
    float gmu = g1 * (1.f / 32.f);
    float grs = rsqrtf(g2 * (1.f / 32.f) - gmu * gmu + kEPS);
#pragma unroll
    for (int k = 0; k < 8; ++k) hreg[k] += (yv[k] - gmu) * grs * wgn[k];
    *(float4*)(hrow)     = *(const float4*)(hreg);
    *(float4*)(hrow + 4) = *(const float4*)(hreg + 4);
  }
}

// =================== FFN: h += (gelu(hn2@Wg_up) * (hn2@Wv_up)) @ Wdown ===================
__launch_bounds__(256)
__global__ void ffn_kernel(float* __restrict__ h,
                           const float* __restrict__ ln2w,
                           const float* __restrict__ wup,   // (E, 2E)
                           const float* __restrict__ wdn) { // (E, E)
  __shared__ _Float16 sBuf[8][16][kE + 8];   // per-wave staging: hn2 then activations
  const int tid  = threadIdx.x;
  const int lane = tid & 31;
  const int w    = tid >> 5;
  const int row  = lane & 15;
  const int half = lane >> 4;
  const size_t rowBase = (size_t)blockIdx.x * 128 + (size_t)w * 16;

  // LN2 -> sBuf (each wave owns 16 rows; 2 lanes per row)
  {
    const float* hp = h + (rowBase + row) * kE + half * 64;
    float v[64] __attribute__((aligned(16)));
#pragma unroll
    for (int k = 0; k < 16; ++k) *(float4*)(v + 4 * k) = *(const float4*)(hp + 4 * k);
    float s1 = 0.f, s2 = 0.f;
#pragma unroll
    for (int k = 0; k < 64; ++k) { s1 += v[k]; s2 += v[k] * v[k]; }
    s1 += __shfl_xor(s1, 16, 32); s2 += __shfl_xor(s2, 16, 32);
    float mu = s1 * (1.f / 128.f);
    float rs = rsqrtf(s2 * (1.f / 128.f) - mu * mu + kEPS);
#pragma unroll
    for (int k = 0; k < 64; ++k)
      sBuf[w][row][half * 64 + k] = (_Float16)((v[k] - mu) * rs * ln2w[half * 64 + k]);
  }

  v16h aH[4];
#pragma unroll
  for (int kf = 0; kf < 4; ++kf)
#pragma unroll
    for (int e = 0; e < 16; ++e)
      aH[kf][e] = sBuf[w][lane & 15][kf * 32 + a_k(lane, e)];

  // up-projection + exact-erf GELU gating, N-tile pairs (gate, value)
  _Float16 actH[8][8];
#pragma unroll
  for (int tt = 0; tt < 8; ++tt) {
    v8f aG = {}, aV = {};
#pragma unroll
    for (int kf = 0; kf < 4; ++kf) {
      v16h bG, bV;
#pragma unroll
      for (int e = 0; e < 16; ++e) {
        int K = kf * 32 + b_k(lane, e);
        int N = lane & 15;
        bG[e] = (_Float16)wup[K * 256 + tt * 16 + N];
        bV[e] = (_Float16)wup[K * 256 + 128 + tt * 16 + N];
      }
      aG = __builtin_amdgcn_wmma_f32_16x16x32_f16(false, aH[kf], false, bG, (short)0, aG, false, false);
      aV = __builtin_amdgcn_wmma_f32_16x16x32_f16(false, aH[kf], false, bV, (short)0, aV, false, false);
    }
#pragma unroll
    for (int k = 0; k < 8; ++k) {
      float g = aG[k];
      actH[tt][k] = (_Float16)(0.5f * g * (1.f + erf_approx(g * 0.70710678f)) * aV[k]);
    }
  }
  // stage activations (overwrite own wave's slice; in-wave LDS ordering suffices)
  {
    int Mb = (lane < 16) ? 0 : 8;
    int N  = lane & 15;
#pragma unroll
    for (int tt = 0; tt < 8; ++tt)
#pragma unroll
      for (int k = 0; k < 8; ++k) sBuf[w][Mb + k][tt * 16 + N] = actH[tt][k];
  }
  v16h aA[4];
#pragma unroll
  for (int kf = 0; kf < 4; ++kf)
#pragma unroll
    for (int e = 0; e < 16; ++e)
      aA[kf][e] = sBuf[w][lane & 15][kf * 32 + a_k(lane, e)];

  // down-projection + residual
#pragma unroll
  for (int tt = 0; tt < 8; ++tt) {
    v8f acc = {};
#pragma unroll
    for (int kf = 0; kf < 4; ++kf) {
      v16h bD;
#pragma unroll
      for (int e = 0; e < 16; ++e)
        bD[e] = (_Float16)wdn[(kf * 32 + b_k(lane, e)) * kE + tt * 16 + (lane & 15)];
      acc = __builtin_amdgcn_wmma_f32_16x16x32_f16(false, aA[kf], false, bD, (short)0, acc, false, false);
    }
    int Mb  = (lane < 16) ? 0 : 8;
    int col = tt * 16 + (lane & 15);
#pragma unroll
    for (int k = 0; k < 8; ++k) {
      size_t idx = (rowBase + Mb + k) * kE + col;
      h[idx] += acc[k];
    }
  }
}

// =================== head: out[b] = LN(h[b,S-1,:],post_w) @ w_out + b_out ===================
__global__ void head_kernel(const float* __restrict__ h, const float* __restrict__ postw,
                            const float* __restrict__ wout, const float* __restrict__ bout,
                            float* __restrict__ out) {
  const int tid = threadIdx.x;
  const int r = tid >> 4, q = tid & 15, c0 = q * 8;
  const size_t b = (size_t)blockIdx.x * 16 + r;
  const float* hp = h + (b * kSEQ + (kSEQ - 1)) * kE + c0;
  float v[8] __attribute__((aligned(16)));
  *(float4*)v       = *(const float4*)hp;
  *(float4*)(v + 4) = *(const float4*)(hp + 4);
  float s1 = 0.f, s2 = 0.f;
#pragma unroll
  for (int k = 0; k < 8; ++k) { s1 += v[k]; s2 += v[k] * v[k]; }
#pragma unroll
  for (int off = 8; off >= 1; off >>= 1) { s1 += __shfl_xor(s1, off, 32); s2 += __shfl_xor(s2, off, 32); }
  float mu = s1 * (1.f / 128.f);
  float rs = rsqrtf(s2 * (1.f / 128.f) - mu * mu + kEPS);
  float dot = 0.f;
#pragma unroll
  for (int k = 0; k < 8; ++k) dot += (v[k] - mu) * rs * postw[c0 + k] * wout[c0 + k];
#pragma unroll
  for (int off = 8; off >= 1; off >>= 1) dot += __shfl_xor(dot, off, 32);
  if (q == 0) out[b] = dot + bout[0];
}

extern "C" void kernel_launch(void* const* d_in, const int* in_sizes, int n_in,
                              void* d_out, int out_size, void* d_ws, size_t ws_size,
                              hipStream_t stream) {
  (void)in_sizes; (void)n_in; (void)out_size; (void)ws_size;
  const float* x     = (const float*)d_in[0];
  const float* w_in  = (const float*)d_in[1];
  const float* b_in  = (const float*)d_in[2];
  const float* ln1_w = (const float*)d_in[3];
  const float* wg    = (const float*)d_in[4];
  const float* rk    = (const float*)d_in[5];
  const float* bcell = (const float*)d_in[6];
  const float* gnw   = (const float*)d_in[7];
  const float* ln2w  = (const float*)d_in[8];
  const float* wup   = (const float*)d_in[9];
  const float* wdn   = (const float*)d_in[10];
  const float* postw = (const float*)d_in[11];
  const float* wout  = (const float*)d_in[12];
  const float* bout  = (const float*)d_in[13];
  float* out = (float*)d_out;
  float* h   = (float*)d_ws;   // B*S*E f32 = 128 MB persistent activations

  int total = kB * kSEQ * kE;
  proj_kernel<<<total / 256, 256, 0, stream>>>(x, w_in, b_in, h, total);
  for (int blk = 0; blk < 2; ++blk) {
    scan_kernel<<<kB / 16, 256, 0, stream>>>(h,
        wg    + (size_t)blk * 4 * kNH * kHD * kHD,
        rk    + (size_t)blk * kNH * kHD * 4 * kHD,
        bcell + (size_t)blk * 4 * kNH * kHD,
        ln1_w + blk * kE, gnw + blk * kE);
    ffn_kernel<<<(kB * kSEQ) / 128, 256, 0, stream>>>(h,
        ln2w + blk * kE,
        wup  + (size_t)blk * kE * 2 * kE,
        wdn  + (size_t)blk * kE * kE);
  }
  head_kernel<<<kB / 16, 256, 0, stream>>>(h, postw, wout, bout, out);
}
